// Dynamics_68642167325163
// MI455X (gfx1250) — matrix operand order repaired and checked by reading, not compiled, not run on hardware
//
#include <hip/hip_runtime.h>
#include <hip/hip_bf16.h>
#include <stdint.h>

typedef __bf16 bf16_t;
typedef __attribute__((ext_vector_type(16))) __bf16 v16bf;
typedef __attribute__((ext_vector_type(8)))  float  v8f;

union FragBF {
  uint4 q[2];
  v16bf v;
};

#define T_STEPS 128
#define HID     512
#define G4H     2048
#define NTHR    512      // 16 waves -> 4 waves/SIMD -> 256-VGPR budget per wave
#define STRIDE_A 1032    // 1024 + 8 bf16 pad -> row stride 2064B, rotates LDS banks

__device__ __forceinline__ float fast_sig(float x)  { return 1.0f / (1.0f + __expf(-x)); }
__device__ __forceinline__ float fast_tanh(float x) { return 1.0f - 2.0f / (__expf(2.0f * x) + 1.0f); }

// ---------------- prep kernels ----------------

__global__ void prep_bias_kernel(const float* __restrict__ b_ih0, const float* __restrict__ b_hh0,
                                 const float* __restrict__ b_ih1, const float* __restrict__ b_hh1,
                                 const float* __restrict__ b_ih2, const float* __restrict__ b_hh2,
                                 float* __restrict__ bsum) {
  int i = blockIdx.x * blockDim.x + threadIdx.x;
  if (i >= 3 * G4H) return;
  int l = i / G4H, n = i % G4H;
  const float* bi = (l == 0) ? b_ih0 : ((l == 1) ? b_ih1 : b_ih2);
  const float* bh = (l == 0) ? b_hh0 : ((l == 1) ? b_hh1 : b_hh2);
  bsum[i] = bi[n] + bh[n];
}

// Re-tile [w_ih; w_hh]^T (K x N) into WMMA-B tiled bf16:
// elem = ((ntile*nkt + kt)*512 + lane*16 + i),
// k = kt*32 + (lane>=16 ? 8 : 0) + (i<8 ? i : 8+i), n = ntile*16 + lane%16.
__global__ void prep_weights_kernel(const float* __restrict__ w_ih,
                                    const float* __restrict__ w_hh,
                                    bf16_t* __restrict__ out,
                                    int d_in, int nkt, int ntiles) {
  int e = blockIdx.x * blockDim.x + threadIdx.x;
  int total = ntiles * nkt * 512;
  if (e >= total) return;
  int ntile = e / (nkt * 512);
  int rem   = e % (nkt * 512);
  int kt    = rem / 512;
  int q     = rem % 512;
  int lane  = q / 16;
  int i     = q % 16;
  int k = kt * 32 + ((lane >= 16) ? 8 : 0) + ((i < 8) ? i : 8 + i);
  int n = ntile * 16 + (lane % 16);
  float val;
  if (k < d_in) val = w_ih[(size_t)n * d_in + k];
  else          val = w_hh[(size_t)n * HID + (k - d_in)];
  out[e] = (bf16_t)val;
}

// ---------------- fragment loads ----------------

__device__ __forceinline__ void load_a(FragBF& f, const bf16_t* ap) {
  const uint4* p = (const uint4*)ap;
  f.q[0] = p[0];
  f.q[1] = p[2];   // second 16B K-run sits +32B per ISA A layout
}

// One k-tile: 8 WMMAs consuming the ring, 8 ring refills at constant immediate
// offsets off the persistent per-lane tile pointers vb[], then one batched
// pointer bump (+1 k-tile) touching no WMMA operand registers (no hazard NOPs).
// Invariant: on entry vb[i] -> tile(i, kt); jj==0 refills slot 7 at kt (+0),
// jj>=1 refill slot jj-1 at kt+1 (+512 elems) => lead stays 7 steps.
__device__ __forceinline__ void body8(const FragBF& a, FragBF (&bq)[8],
                                      const bf16_t* (&vb)[8], v8f* acc) {
#pragma unroll
  for (int jj = 0; jj < 8; ++jj) {
    const int slot = (jj + 7) & 7;
    const uint4* p = (const uint4*)(vb[slot] + (jj == 0 ? 0 : 512));
    bq[slot].q[0] = p[0];
    bq[slot].q[1] = p[1];
    __builtin_prefetch((const void*)(vb[jj] + 1024), 0, 3);   // 2 k-tiles ahead
    acc[jj] = __builtin_amdgcn_wmma_f32_16x16x32_bf16(
        false, a.v, false, bq[jj].v, (short)0, acc[jj], false, false);
  }
#pragma unroll
  for (int i = 0; i < 8; ++i) vb[i] += 512;
}

// Software-pipelined 16xK x Kx(8x16) GEMM. A fragments double-buffered one full
// body (8 WMMAs) ahead via explicit a0/a1 pairing (all indices compile-time);
// trailing ring refills overrun by <= 1 tile but stay inside the workspace.
template <int NKT>
__device__ __forceinline__ void gemm8(const bf16_t* aBase, const bf16_t* W,
                                      const int* boff, v8f* acc) {
  FragBF a0, a1;
  FragBF bq[8];
  const bf16_t* vb[8];
#pragma unroll
  for (int i = 0; i < 8; ++i) vb[i] = W + boff[i];
#pragma unroll
  for (int i = 0; i < 7; ++i) {            // prologue: slots 0..6 at kt=0
    const uint4* p = (const uint4*)(vb[i]);
    bq[i].q[0] = p[0];
    bq[i].q[1] = p[1];
  }
  const bf16_t* ap = aBase;
  load_a(a0, ap); ap += 32;
#pragma unroll 1
  for (int g = 0; g < NKT / 2; ++g) {
    load_a(a1, ap); ap += 32;              // A for the next body, 8 WMMAs early
    body8(a0, bq, vb, acc);
    load_a(a0, ap); ap += 32;
    body8(a1, bq, vb, acc);
  }
  if (NKT & 1)
    body8(a0, bq, vb, acc);
}

// ---------------- main LSTM kernel ----------------
// grid = 16 blocks (16 batch rows each), 512 threads = 16 waves (wave32).
// Wave w owns n-tiles {j*32 + 2w + p : j=0..3, p=0..1} = i/f/g/o for hidden 32w..32w+31.

__global__ __launch_bounds__(NTHR, 1) void lstm3_kernel(
    const float* __restrict__ state, const float* __restrict__ action,
    const float* __restrict__ bsum,
    const bf16_t* __restrict__ w0, const bf16_t* __restrict__ w1,
    const bf16_t* __restrict__ w2, const bf16_t* __restrict__ wlin,
    const float* __restrict__ lin_b,
    bf16_t* __restrict__ xb0, bf16_t* __restrict__ xb1,
    float* __restrict__ out) {
  __shared__ __align__(16) bf16_t sA[17 * STRIDE_A];   // +1 pad row: trailing A loads stay in-bounds

  const int tid  = threadIdx.x;
  const int lane = tid & 31;
  const int wave = tid >> 5;        // 0..15
  const int blk  = blockIdx.x;
  const int half = (lane >= 16) ? 1 : 0;
  const int ncol = lane & 15;

  // per-lane A base (row = lane%16, first K-run offset = half*8) -- layout-invariant
  const bf16_t* aBase = sA + (lane & 15) * STRIDE_A + half * 8;

  for (int layer = 0; layer < 3; ++layer) {
    const int d_in = (layer == 0) ? 96 : 512;
    const int nkt  = (d_in + HID) / 32;
    const bf16_t* W   = (layer == 0) ? w0 : ((layer == 1) ? w1 : w2);
    const bf16_t* xin = (layer == 1) ? xb0 : xb1;   // unused for layer 0
    bf16_t* xout      = (layer == 0) ? xb0 : xb1;   // unused for layer 2

    // per-wave W tile offsets: jj = j*2 + p -> ntile = j*32 + 2*wave + p
    int boff[8];
    float bias[8];
#pragma unroll
    for (int jj = 0; jj < 8; ++jj) {
      int nt = (jj >> 1) * 32 + 2 * wave + (jj & 1);
      boff[jj] = nt * nkt * 512 + lane * 16;
      bias[jj] = bsum[layer * G4H + (jj >> 1) * 512 + 32 * wave + (jj & 1) * 16 + ncol];
    }

    float cst[2][8];
#pragma unroll
    for (int p = 0; p < 2; ++p)
#pragma unroll
      for (int r = 0; r < 8; ++r) cst[p][r] = 0.0f;

    // h_{-1} = 0
    for (int idx = tid; idx < 16 * HID; idx += NTHR) {
      int row = idx >> 9, col = idx & 511;
      sA[row * STRIDE_A + d_in + col] = (bf16_t)0.0f;
    }
    // stage x_0
    if (layer == 0) {
      for (int idx = tid; idx < 16 * 96; idx += NTHR) {
        int row = idx / 96, col = idx % 96;
        int gr = blk * 16 + row;
        float v = (col < 64) ? state[((size_t)gr * T_STEPS) * 64 + col]
                             : action[((size_t)gr * T_STEPS) * 32 + (col - 64)];
        sA[row * STRIDE_A + col] = (bf16_t)v;
      }
    } else {
#pragma unroll
      for (int kk = 0; kk < 2; ++kk) {
        int u = tid + kk * NTHR;
        int row = u >> 6, seg = u & 63;
        const bf16_t* src = xin + (((size_t)(blk * 16 + row)) * T_STEPS) * HID + seg * 8;
        uint64_t ga = (uint64_t)(uintptr_t)src;
        uint32_t lo = (uint32_t)(uintptr_t)(sA + row * STRIDE_A) + (uint32_t)(seg * 16);
        asm volatile("global_load_async_to_lds_b128 %0, %1, off" :: "v"(lo), "v"(ga) : "memory");
      }
    }
    asm volatile("s_wait_asynccnt 0x0" ::: "memory");
    __syncthreads();

    for (int t = 0; t < T_STEPS; ++t) {
      v8f acc[8];
#pragma unroll
      for (int jj = 0; jj < 8; ++jj)
#pragma unroll
        for (int r = 0; r < 8; ++r) acc[jj][r] = bias[jj];

      if (layer == 0)
        gemm8<19>(aBase, W, boff, acc);
      else
        gemm8<32>(aBase, W, boff, acc);

      __syncthreads();   // all waves done reading x_t / h_{t-1}

#pragma unroll
      for (int p = 0; p < 2; ++p) {
        const int jcol = 32 * wave + 16 * p + ncol;   // hidden unit
#pragma unroll
        for (int r = 0; r < 8; ++r) {
          float iv = fast_sig(acc[0 + p][r]);
          float fv = fast_sig(acc[2 + p][r]);
          float gv = fast_tanh(acc[4 + p][r]);
          float ov = fast_sig(acc[6 + p][r]);
          float cv = fv * cst[p][r] + iv * gv;
          cst[p][r] = cv;
          float hv = ov * fast_tanh(cv);
          bf16_t hb = (bf16_t)hv;
          int row = r + half * 8;
          sA[row * STRIDE_A + d_in + jcol] = hb;
          if (layer < 2)
            xout[(((size_t)(blk * 16 + row)) * T_STEPS + t) * HID + jcol] = hb;
        }
      }
      // stage x_{t+1}
      if (t + 1 < T_STEPS) {
        if (layer == 0) {
          for (int idx = tid; idx < 16 * 96; idx += NTHR) {
            int row = idx / 96, col = idx % 96;
            int gr = blk * 16 + row;
            float v = (col < 64) ? state[((size_t)gr * T_STEPS + (t + 1)) * 64 + col]
                                 : action[((size_t)gr * T_STEPS + (t + 1)) * 32 + (col - 64)];
            sA[row * STRIDE_A + col] = (bf16_t)v;
          }
        } else {
#pragma unroll
          for (int kk = 0; kk < 2; ++kk) {
            int u = tid + kk * NTHR;
            int row = u >> 6, seg = u & 63;
            const bf16_t* src = xin + (((size_t)(blk * 16 + row)) * T_STEPS + (t + 1)) * HID + seg * 8;
            uint64_t ga = (uint64_t)(uintptr_t)src;
            uint32_t lo = (uint32_t)(uintptr_t)(sA + row * STRIDE_A) + (uint32_t)(seg * 16);
            asm volatile("global_load_async_to_lds_b128 %0, %1, off" :: "v"(lo), "v"(ga) : "memory");
          }
        }
      }
      asm volatile("s_wait_asynccnt 0x0" ::: "memory");
      __syncthreads();
    }
  }

  // Final linear: out[16,64] = h_last(bf16, in sA h-region) x wlin + lin_b
  if (wave < 4) {
    v8f acc;
    float bo = lin_b[wave * 16 + ncol];
#pragma unroll
    for (int r = 0; r < 8; ++r) acc[r] = bo;
    const bf16_t* aB = aBase + 512;   // h-region (d_in = 512 for layer 2)
    const bf16_t* bB = wlin + ((size_t)(wave * 16)) * 512 + lane * 16;
    FragBF a, b;
#pragma unroll
    for (int kt = 0; kt < 16; ++kt) {
      load_a(a, aB + kt * 32);
      const uint4* p = (const uint4*)(bB + (size_t)kt * 512);
      b.q[0] = p[0];
      b.q[1] = p[1];
      acc = __builtin_amdgcn_wmma_f32_16x16x32_bf16(
          false, a.v, false, b.v, (short)0, acc, false, false);
    }
#pragma unroll
    for (int r = 0; r < 8; ++r) {
      int row = blk * 16 + r + half * 8;
      out[(size_t)row * 64 + wave * 16 + ncol] = acc[r];
    }
  }
}

// ---------------- launch ----------------

extern "C" void kernel_launch(void* const* d_in, const int* in_sizes, int n_in,
                              void* d_out, int out_size, void* d_ws, size_t ws_size,
                              hipStream_t stream) {
  (void)in_sizes; (void)n_in; (void)out_size; (void)ws_size;
  const float* state  = (const float*)d_in[0];
  const float* action = (const float*)d_in[1];
  const float* w_ih0 = (const float*)d_in[2];
  const float* w_hh0 = (const float*)d_in[3];
  const float* b_ih0 = (const float*)d_in[4];
  const float* b_hh0 = (const float*)d_in[5];
  const float* w_ih1 = (const float*)d_in[6];
  const float* w_hh1 = (const float*)d_in[7];
  const float* b_ih1 = (const float*)d_in[8];
  const float* b_hh1 = (const float*)d_in[9];
  const float* w_ih2 = (const float*)d_in[10];
  const float* w_hh2 = (const float*)d_in[11];
  const float* b_ih2 = (const float*)d_in[12];
  const float* b_hh2 = (const float*)d_in[13];
  const float* lin_w = (const float*)d_in[14];
  const float* lin_b = (const float*)d_in[15];
  float* out = (float*)d_out;

  char* ws = (char*)d_ws;
  size_t off = 0;
  float* bsum = (float*)(ws + off); off += (size_t)3 * G4H * sizeof(float);
  off = (off + 255) & ~(size_t)255;
  bf16_t* w0 = (bf16_t*)(ws + off); off += (size_t)128 * 19 * 512 * 2;
  bf16_t* w1 = (bf16_t*)(ws + off); off += (size_t)128 * 32 * 512 * 2;
  bf16_t* w2 = (bf16_t*)(ws + off); off += (size_t)128 * 32 * 512 * 2;
  bf16_t* wl = (bf16_t*)(ws + off); off += (size_t)4 * 16 * 512 * 2;
  bf16_t* xb0 = (bf16_t*)(ws + off); off += (size_t)256 * T_STEPS * HID * 2;
  bf16_t* xb1 = (bf16_t*)(ws + off); off += (size_t)256 * T_STEPS * HID * 2;

  prep_bias_kernel<<<(3 * G4H + 255) / 256, 256, 0, stream>>>(
      b_ih0, b_hh0, b_ih1, b_hh1, b_ih2, b_hh2, bsum);
  {
    int total = 128 * 19 * 512;
    prep_weights_kernel<<<(total + 255) / 256, 256, 0, stream>>>(w_ih0, w_hh0, w0, 96, 19, 128);
  }
  {
    int total = 128 * 32 * 512;
    prep_weights_kernel<<<(total + 255) / 256, 256, 0, stream>>>(w_ih1, w_hh1, w1, 512, 32, 128);
    prep_weights_kernel<<<(total + 255) / 256, 256, 0, stream>>>(w_ih2, w_hh2, w2, 512, 32, 128);
  }
  {
    int total = 4 * 16 * 512;
    prep_weights_kernel<<<(total + 255) / 256, 256, 0, stream>>>(lin_w, lin_w, wl, 512, 16, 4);
  }
  lstm3_kernel<<<16, NTHR, 0, stream>>>(state, action, bsum, w0, w1, w2, wl, lin_b, xb0, xb1, out);
}
